// HL_HGCNN_zinc_dense_int3_pyr_74680891342975
// MI455X (gfx1250) — matrix operand order repaired and checked by reading, not compiled, not run on hardware
//
#include <hip/hip_runtime.h>
#include <hip/hip_bf16.h>
#include <stdint.h>

// ---------------------------------------------------------------------------
// HL-HGCNN (Hodge-Laguerre hypergraph GCN) forward pass for MI455X (gfx1250).
// Dense GEMMs: v_wmma_f32_16x16x32_f16 (wave32 WMMA), f32 accumulate, fused
// bias/accumulate/ReLU epilogue, double-buffered LDS tiles, async global->LDS
// (GLOBAL_LOAD_ASYNC_TO_LDS_B64, ASYNCcnt-tracked) for the A tile.
// Sparse segment ops use f32 atomics (accumulators sized for the 192MB L2).
// ---------------------------------------------------------------------------

typedef _Float16 half_t;
typedef __attribute__((ext_vector_type(16))) _Float16 v16h;
typedef __attribute__((ext_vector_type(8)))  float    v8f;
typedef __attribute__((ext_vector_type(2)))  int      v2i_ty;

#define N_NODE   30000
#define N_EDGE   60000
#define G_BATCH  1024
#define NNZ_T    (N_NODE + 4 * N_EDGE)   // 270000
#define NNZ_S    (5 * N_EDGE)            // 300000
#define LD0      1984                    // final concat width (row stride of x_t0/x_s0)

// GEMM tiling: 256 threads = 8 wave32; tile 128(M) x 64(N), K-step 32.
#define TM 128
#define TN 64
#define TK 32
#define LDS_STRIDE 36   // 72B row stride: 8B-aligned fills, conflict-free frag reads

#define AS1 __attribute__((address_space(1)))
#define AS3 __attribute__((address_space(3)))

// ---- optional CDNA5 async global->LDS path (compile-guarded) ----
#if __has_builtin(__builtin_amdgcn_global_load_async_to_lds_b64)
#define HAVE_ASYNC_LDS 1
#else
#define HAVE_ASYNC_LDS 0
#endif

#if HAVE_ASYNC_LDS
#if __has_builtin(__builtin_amdgcn_s_wait_asynccnt)
#define WAIT_ASYNC() __builtin_amdgcn_s_wait_asynccnt(0)
#else
#define WAIT_ASYNC() asm volatile("s_wait_asynccnt 0" ::: "memory")
#endif
#else
#define WAIT_ASYNC() ((void)0)
#endif

// ============================ WMMA GEMM ====================================
// C[M,N](f32) = A[M,K](f16,row-major,lda) x B[K,N](f16,row-major,ldb)
//             (+ bias[col]) (+ C prior if flags&1) (ReLU if flags&2)
__launch_bounds__(256)
__global__ void gemm_f16_wmma(const half_t* __restrict__ A, int lda,
                              const half_t* __restrict__ B, int ldb,
                              float* __restrict__ C, int ldc,
                              const float* __restrict__ bias,
                              int M, int N, int K, int flags)
{
    __shared__ half_t As[2][TM * LDS_STRIDE];
    __shared__ half_t Bs[2][TN * LDS_STRIDE];

    const int tid  = threadIdx.x;
    const int lane = tid & 31;
    const int wave = tid >> 5;
    const int wm   = wave & 3;     // 4 waves along M (32 rows each)
    const int wn   = wave >> 2;    // 2 waves along N (32 cols each)
    const int m0   = blockIdx.y * TM;
    const int n0   = blockIdx.x * TN;
    const bool fullM = (m0 + TM <= M);

    v8f vz = {};
    v8f acc[2][2];
    acc[0][0] = vz; acc[0][1] = vz; acc[1][0] = vz; acc[1][1] = vz;

    const int ar  = tid >> 1;          // A-tile row 0..127 (manual path)
    const int ah  = (tid & 1) * 16;    // A-tile col half (0 or 16)
    const int bnc = tid & 63;          // B-tile col 0..63
    const int bq  = (tid >> 6) * 8;    // B-tile k start (0,8,16,24)

    // ---- tile loaders (issue only; completion handled at the sync point) ----
    auto loadA = [&](int k0, int buf) {
        if (fullM && (k0 + TK) <= K) {
#if HAVE_ASYNC_LDS
            // 128 rows x 64B = 1024 x b64 chunks (8B-aligned at both ends for
            // the 72B LDS row stride); 4 async DMA issues per thread.
#pragma unroll
            for (int c = tid; c < TM * 8; c += 256) {
                int r = c >> 3, q = (c & 7) * 4;   // q in halves: 0,4,...,28
                __builtin_amdgcn_global_load_async_to_lds_b64(
                    (AS1 v2i_ty*)(A + (size_t)(m0 + r) * lda + k0 + q),
                    (AS3 v2i_ty*)(&As[buf][r * LDS_STRIDE + q]),
                    0, 0);
            }
#else
            const uint32_t* srcp = (const uint32_t*)(A + (size_t)(m0 + ar) * lda + k0 + ah);
            uint32_t* d32 = (uint32_t*)&As[buf][ar * LDS_STRIDE + ah];
#pragma unroll
            for (int i = 0; i < 8; ++i) d32[i] = srcp[i];
#endif
        } else {
            // branch-free edge path: clamped loads + select-zero (v_cndmask, no exec juggling)
            int gr  = m0 + ar;
            int grc = (gr < M) ? gr : 0;
            half_t* dstp = &As[buf][ar * LDS_STRIDE + ah];
#pragma unroll
            for (int i = 0; i < 16; ++i) {
                int kk  = k0 + ah + i;
                int kkc = (kk < K) ? kk : 0;
                half_t v = A[(size_t)grc * lda + kkc];
                dstp[i] = (gr < M && kk < K) ? v : (half_t)0.0f;
            }
        }
    };
    auto loadB = [&](int k0, int buf) {
        // stored transposed: Bs[n][k] so K-pairs are contiguous for frag loads
        int gn = n0 + bnc;
        half_t* dstp = &Bs[buf][bnc * LDS_STRIDE + bq];
        if (gn < N && (k0 + TK) <= K) {
            const half_t* gp = B + (size_t)(k0 + bq) * ldb + gn;
#pragma unroll
            for (int i = 0; i < 8; ++i) dstp[i] = gp[(size_t)i * ldb];
        } else {
            int gnc = (gn < N) ? gn : 0;
#pragma unroll
            for (int i = 0; i < 8; ++i) {
                int kk  = k0 + bq + i;
                int kkc = (kk < K) ? kk : 0;
                half_t v = B[(size_t)kkc * ldb + gnc];
                dstp[i] = (gn < N && kk < K) ? v : (half_t)0.0f;
            }
        }
    };

    union frag { uint32_t u[8]; v16h v; };

    // ---- double-buffered main loop ----
    int buf = 0;
    loadA(0, 0);
    loadB(0, 0);
    WAIT_ASYNC();
    __syncthreads();

    for (int k0 = 0; k0 < K; k0 += TK) {
        int nk = k0 + TK;
        if (nk < K) { loadA(nk, buf ^ 1); loadB(nk, buf ^ 1); }

        // B fragments (hoisted; shared across both A fragments):
        // lane<16 holds K 0..7,16..23 of col (lane&15); lane>=16: +8
        frag bfr[2];
#pragma unroll
        for (int fn = 0; fn < 2; ++fn) {
            const half_t* row = &Bs[buf][(wn * 32 + fn * 16 + (lane & 15)) * LDS_STRIDE];
            int kg = (lane >> 4) * 8;
#pragma unroll
            for (int q = 0; q < 8; ++q)
                bfr[fn].u[q] = *(const uint32_t*)(row + ((q < 4) ? (2 * q) : (16 + 2 * (q - 4))) + kg);
        }
#pragma unroll
        for (int fm = 0; fm < 2; ++fm) {
            // A fragment per ISA 16-bit 16x32 layout:
            // lane<16: VGPR0-3 -> K 0..7, VGPR4-7 -> K 16..23 ; lane>=16: +8
            frag afr;
            {
                const half_t* row = &As[buf][(wm * 32 + fm * 16 + (lane & 15)) * LDS_STRIDE];
                int kh = (lane >> 4) * 8;
#pragma unroll
                for (int q = 0; q < 8; ++q) {
                    int kb = ((q < 4) ? (2 * q) : (16 + 2 * (q - 4))) + kh;
                    afr.u[q] = *(const uint32_t*)(row + kb);
                }
            }
#pragma unroll
            for (int fn = 0; fn < 2; ++fn)
                acc[fm][fn] = __builtin_amdgcn_wmma_f32_16x16x32_f16(
                    false, afr.v, false, bfr[fn].v, (short)0, acc[fm][fn], false, false);
        }

        WAIT_ASYNC();          // next tile's async DMA done before the barrier
        __syncthreads();
        buf ^= 1;
    }

    // ---- epilogue: D layout VGPR q -> row q + 8*(lane>=16), col lane&15 ----
    const int kg = lane >> 4;
    const int nl = lane & 15;
#pragma unroll
    for (int fm = 0; fm < 2; ++fm) {
#pragma unroll
        for (int fn = 0; fn < 2; ++fn) {
            int col = n0 + wn * 32 + fn * 16 + nl;
            if (col >= N) continue;
            float bv = bias ? bias[col] : 0.0f;
#pragma unroll
            for (int q = 0; q < 8; ++q) {
                int row = m0 + wm * 32 + fm * 16 + q + 8 * kg;
                if (row >= M) continue;
                size_t off = (size_t)row * ldc + col;
                float val = acc[fm][fn][q] + bv;
                if (flags & 1) val += C[off];
                if (flags & 2) val = fmaxf(val, 0.0f);
                C[off] = val;
            }
        }
    }
}

// ========================= elementwise / sparse kernels =====================
__global__ void fill0_kernel(float* p, long n) {
    long i = (long)blockIdx.x * blockDim.x + threadIdx.x;
    if (i < n) p[i] = 0.0f;
}
__global__ void cast_h_kernel(const float* __restrict__ s, half_t* __restrict__ d, long n) {
    long i = (long)blockIdx.x * blockDim.x + threadIdx.x;
    if (i < n) d[i] = (half_t)s[i];
}
// out[r] = concat(emb[int(x[r,0])] (57), x[r,1:8] (7)) -> 64 wide
__global__ void embed_kernel(const float* __restrict__ x, const float* __restrict__ emb,
                             float* __restrict__ out, int R) {
    long i = (long)blockIdx.x * blockDim.x + threadIdx.x;
    if (i >= (long)R * 64) return;
    int r = (int)(i >> 6), c = (int)(i & 63);
    float v;
    if (c < 57) { int a = (int)x[(size_t)r * 8]; v = emb[a * 57 + c]; }
    else        { v = x[(size_t)r * 8 + (c - 57) + 1]; }
    out[i] = v;
}
__global__ void deg_kernel(const int* __restrict__ src, const int* __restrict__ dst,
                           float* deg, int E) {
    long i = (long)blockIdx.x * blockDim.x + threadIdx.x;
    if (i >= 2L * E) return;
    int n = (i < E) ? src[i] : dst[i - E];
    atomicAdd(&deg[n], 1.0f);
}
__global__ void clamp1_kernel(float* p, int n) {
    int i = blockIdx.x * blockDim.x + threadIdx.x;
    if (i < n) p[i] = fmaxf(p[i], 1.0f);
}
// agg[src[e]] += z[e]; agg[dst[e]] += z[e]
__global__ void scatter2_kernel(const float* __restrict__ z, const int* __restrict__ src,
                                const int* __restrict__ dst, float* agg, int E, int F) {
    long i = (long)blockIdx.x * blockDim.x + threadIdx.x;
    if (i >= (long)E * F) return;
    int e = (int)(i / F), c = (int)(i - (long)e * F);
    float v = z[i];
    atomicAdd(&agg[(long)src[e] * F + c], v);
    atomicAdd(&agg[(long)dst[e] * F + c], v);
}
// x_t = relu(tmp + agg/deg)   (writes f32 + f16 copies)
__global__ void combine_node_kernel(const float* __restrict__ tmp, const float* __restrict__ agg,
                                    const float* __restrict__ deg, float* xf, half_t* xh,
                                    int R, int F) {
    long i = (long)blockIdx.x * blockDim.x + threadIdx.x;
    if (i >= (long)R * F) return;
    int r = (int)(i / F);
    float v = fmaxf(tmp[i] + agg[i] / deg[r], 0.0f);
    xf[i] = v; xh[i] = (half_t)v;
}
// x_s = relu(tmp + 0.5*(zt[src]+zt[dst]))
__global__ void combine_edge_kernel(const float* __restrict__ tmp, const float* __restrict__ zt,
                                    const int* __restrict__ src, const int* __restrict__ dst,
                                    float* xf, half_t* xh, int E, int F) {
    long i = (long)blockIdx.x * blockDim.x + threadIdx.x;
    if (i >= (long)E * F) return;
    int e = (int)(i / F), c = (int)(i - (long)e * F);
    float v = tmp[i] + 0.5f * (zt[(long)src[e] * F + c] + zt[(long)dst[e] * F + c]);
    v = fmaxf(v, 0.0f);
    xf[i] = v; xh[i] = (half_t)v;
}
// out[rows[e]] += w[e] * x[cols[e]]
__global__ void spmm_kernel(const int* __restrict__ rows, const int* __restrict__ cols,
                            const float* __restrict__ w, const float* __restrict__ x,
                            float* out, int nnz, int F) {
    long i = (long)blockIdx.x * blockDim.x + threadIdx.x;
    if (i >= (long)nnz * F) return;
    int e = (int)(i / F), c = (int)(i - (long)e * F);
    atomicAdd(&out[(long)rows[e] * F + c], w[e] * x[(long)cols[e] * F + c]);
}
// u = f16(x - lap)   (Laguerre term (I - L) x)
__global__ void subcast_kernel(const float* __restrict__ x, const float* __restrict__ lap,
                               half_t* u, long n) {
    long i = (long)blockIdx.x * blockDim.x + threadIdx.x;
    if (i < n) u[i] = (half_t)(x[i] - lap[i]);
}
// stats[c] += x ; stats[F+c] += x*x
__global__ void colstat_kernel(const float* __restrict__ x, float* stats, int R, int F) {
    long i = (long)blockIdx.x * blockDim.x + threadIdx.x;
    if (i >= (long)R * F) return;
    int c = (int)(i % F);
    float v = x[i];
    atomicAdd(&stats[c], v);
    atomicAdd(&stats[F + c], v * v);
}
// y = relu((x-m)*rsqrt(var+eps)*g + be); writes f32 (ld=F) and f16 (ld=ldh)
__global__ void bn_relu_kernel(const float* __restrict__ x, const float* __restrict__ stats,
                               const float* __restrict__ g, const float* __restrict__ be,
                               float* outf, half_t* outh, int ldh, int R, int F) {
    long i = (long)blockIdx.x * blockDim.x + threadIdx.x;
    if (i >= (long)R * F) return;
    int c = (int)(i % F);
    float m  = stats[c] / (float)R;
    float vr = stats[F + c] / (float)R - m * m;
    float y  = (x[i] - m) * rsqrtf(vr + 1e-5f) * g[c] + be[c];
    y = fmaxf(y, 0.0f);
    outf[i] = y;
    outh[(i / F) * (long)ldh + c] = (half_t)y;
}
__global__ void count_kernel(const int* __restrict__ seg, float* cnt, int R) {
    long i = (long)blockIdx.x * blockDim.x + threadIdx.x;
    if (i < R) atomicAdd(&cnt[seg[i]], 1.0f);
}
__global__ void pool_accum_kernel(const float* __restrict__ x, const int* __restrict__ seg,
                                  float* pooled, int R, int F, int colOff) {
    long i = (long)blockIdx.x * blockDim.x + threadIdx.x;
    if (i >= (long)R * F) return;
    int r = (int)(i / F), c = (int)(i - (long)r * F);
    atomicAdd(&pooled[(long)seg[r] * 1024 + colOff + c], x[i]);
}
// out[g] = bias + sum_c pooled[g,c]/count * W[c]   (pooled = [edge 512 | node 512])
__global__ void out_kernel(const float* __restrict__ pooled, const float* __restrict__ cnt,
                           const float* __restrict__ W, const float* __restrict__ b,
                           float* out, int Gn) {
    int g = blockIdx.x * blockDim.x + threadIdx.x;
    if (g >= Gn) return;
    float cs = fmaxf(cnt[g], 1.0f);
    float ct = fmaxf(cnt[Gn + g], 1.0f);
    float acc = b[0];
    const float* row = pooled + (long)g * 1024;
    for (int c = 0; c < 512; ++c)  acc += row[c] / cs * W[c];
    for (int c = 0; c < 512; ++c)  acc += row[512 + c] / ct * W[512 + c];
    out[g] = acc;
}

// ============================== host orchestration ==========================
static inline dim3 g1(long n) { return dim3((unsigned)((n + 255) / 256)); }

extern "C" void kernel_launch(void* const* d_in, const int* in_sizes, int n_in,
                              void* d_out, int out_size, void* d_ws, size_t ws_size,
                              hipStream_t stream)
{
    (void)in_sizes; (void)n_in; (void)out_size; (void)ws_size;

    // -------- inputs (setup_inputs() dict insertion order) --------
    const float* x_t_in  = (const float*)d_in[0];   // (30000,8)
    const float* x_s_in  = (const float*)d_in[1];   // (60000,8)
    const int*   ei      = (const int*)d_in[2];     // (2,E): src=ei, dst=ei+E
    const int*   ei_t    = (const int*)d_in[3];     // (2,NNZ_T)
    const float* w_t     = (const float*)d_in[4];
    const int*   ei_s    = (const int*)d_in[5];     // (2,NNZ_S)
    const float* w_s     = (const float*)d_in[6];
    const int*   n_batch = (const int*)d_in[7];
    const int*   s_batch = (const int*)d_in[8];
    // params flattened leaf-by-leaf in nested insertion order:
    // 9 node_emb; 10/11 init_t.{W,b}; 12/13 init_bn_t.{g,be};
    // 14/15 init_s.{W,b}; 16/17 init_bn_s.{g,be};
    // per block (16 leaves): t.{W,b} s.{W,b} ts.{W,b} st.{W,b}
    //                        conv_t.{W,b} bn_t.{g,be} conv_s.{W,b} bn_s.{g,be}
    // 146/147 out.{W,b}
    auto P = [&](int i) -> const float* { return (const float*)d_in[i]; };
    const int* src = ei;
    const int* dst = ei + N_EDGE;

    // -------- workspace partition --------
    size_t off = 0;
    auto alloc = [&](size_t bytes) -> void* {
        void* p = (char*)d_ws + off;
        off += (bytes + 255) & ~(size_t)255;
        return p;
    };
    half_t* xt0   = (half_t*)alloc((size_t)N_NODE * LD0 * sizeof(half_t)); // concat bank (nodes)
    half_t* xs0   = (half_t*)alloc((size_t)N_EDGE * LD0 * sizeof(half_t)); // concat bank (edges)
    float*  xt_f  = (float*) alloc((size_t)N_NODE * 512 * sizeof(float));
    float*  xs_f  = (float*) alloc((size_t)N_EDGE * 512 * sizeof(float));
    half_t* xt_h  = (half_t*)alloc((size_t)N_NODE * 512 * sizeof(half_t));
    half_t* xs_h  = (half_t*)alloc((size_t)N_EDGE * 512 * sizeof(half_t));
    half_t* u_h   = (half_t*)alloc((size_t)N_EDGE * 512 * sizeof(half_t)); // max(N,E) rows
    float*  zt    = (float*) alloc((size_t)N_NODE * 512 * sizeof(float));
    float*  zs    = (float*) alloc((size_t)N_EDGE * 512 * sizeof(float));
    float*  agg   = (float*) alloc((size_t)N_NODE * 512 * sizeof(float));
    float*  tmp_t = (float*) alloc((size_t)N_NODE * 512 * sizeof(float));
    float*  tmp_s = (float*) alloc((size_t)N_EDGE * 512 * sizeof(float));
    float*  lap   = (float*) alloc((size_t)N_EDGE * 512 * sizeof(float));
    float*  deg   = (float*) alloc((size_t)N_NODE * sizeof(float));
    float*  stats = (float*) alloc(1024 * sizeof(float));                  // [sum|sumsq] up to F=512
    float*  pooled= (float*) alloc((size_t)G_BATCH * 1024 * sizeof(float));
    float*  cnt   = (float*) alloc(2 * G_BATCH * sizeof(float));
    half_t* wbuf  = (half_t*)alloc((size_t)1600000 * sizeof(half_t));      // f16 weight staging

    auto gemm = [&](const half_t* A, int lda, const half_t* B, int n, float* C, int ldc,
                    const float* bias, int m, int k, int flags) {
        dim3 grid((n + TN - 1) / TN, (m + TM - 1) / TM);
        gemm_f16_wmma<<<grid, 256, 0, stream>>>(A, lda, B, n, C, ldc, bias, m, n, k, flags);
    };
    auto castw = [&](const float* s, long n, half_t* d) {
        cast_h_kernel<<<g1(n), 256, 0, stream>>>(s, d, n);
    };
    auto fill0 = [&](float* p, long n) { fill0_kernel<<<g1(n), 256, 0, stream>>>(p, n); };

    // ================= embedding + init K=1 HL-convs =================
    embed_kernel<<<g1((long)N_NODE * 64), 256, 0, stream>>>(x_t_in, P(9), tmp_t, N_NODE);
    cast_h_kernel<<<g1((long)N_NODE * 64), 256, 0, stream>>>(tmp_t, xt_h, (long)N_NODE * 64);
    castw(P(10), 64L * 64, wbuf);
    gemm(xt_h, 64, wbuf, 64, tmp_t, 64, P(11), N_NODE, 64, 0);
    fill0(stats, 1024);
    colstat_kernel<<<g1((long)N_NODE * 64), 256, 0, stream>>>(tmp_t, stats, N_NODE, 64);
    bn_relu_kernel<<<g1((long)N_NODE * 64), 256, 0, stream>>>(tmp_t, stats, P(12), P(13),
                                                              xt_f, xt0, LD0, N_NODE, 64);

    embed_kernel<<<g1((long)N_EDGE * 64), 256, 0, stream>>>(x_s_in, P(9), tmp_s, N_EDGE);
    cast_h_kernel<<<g1((long)N_EDGE * 64), 256, 0, stream>>>(tmp_s, xs_h, (long)N_EDGE * 64);
    castw(P(14), 64L * 64, wbuf);
    gemm(xs_h, 64, wbuf, 64, tmp_s, 64, P(15), N_EDGE, 64, 0);
    fill0(stats, 1024);
    colstat_kernel<<<g1((long)N_EDGE * 64), 256, 0, stream>>>(tmp_s, stats, N_EDGE, 64);
    bn_relu_kernel<<<g1((long)N_EDGE * 64), 256, 0, stream>>>(tmp_s, stats, P(16), P(17),
                                                              xs_f, xs0, LD0, N_EDGE, 64);

    // degree
    fill0(deg, N_NODE);
    deg_kernel<<<g1(2L * N_EDGE), 256, 0, stream>>>(src, dst, deg, N_EDGE);
    clamp1_kernel<<<g1(N_NODE), 256, 0, stream>>>(deg, N_NODE);

    // ================= dense pyramid blocks =================
    static const int DIN[8] = {64, 128, 192, 320, 448, 704, 960, 1472};
    static const int FF[8]  = {64, 64, 128, 128, 256, 256, 512, 512};

    for (int blk = 0; blk < 8; ++blk) {
        const int din = DIN[blk], f = FF[blk];
        const int pb = 18 + 16 * blk;
        const long Nf = (long)N_NODE * f, Ef = (long)N_EDGE * f;

        // z_s = x_s0 @ W_st + b_st
        castw(P(pb + 6), (long)din * f, wbuf);
        gemm(xs0, LD0, wbuf, f, zs, f, P(pb + 7), N_EDGE, din, 0);
        // agg_t = scatter(z_s -> src) + scatter(z_s -> dst)
        fill0(agg, Nf);
        scatter2_kernel<<<g1(Ef), 256, 0, stream>>>(zs, src, dst, agg, N_EDGE, f);
        // x_t = relu(x_t0 @ W_t + b_t + agg/deg)
        castw(P(pb + 0), (long)din * f, wbuf);
        gemm(xt0, LD0, wbuf, f, tmp_t, f, P(pb + 1), N_NODE, din, 0);
        combine_node_kernel<<<g1(Nf), 256, 0, stream>>>(tmp_t, agg, deg, xt_f, xt_h, N_NODE, f);
        // z_t = x_t0 @ W_ts + b_ts
        castw(P(pb + 4), (long)din * f, wbuf);
        gemm(xt0, LD0, wbuf, f, zt, f, P(pb + 5), N_NODE, din, 0);
        // x_s = relu(x_s0 @ W_s + b_s + 0.5*(z_t[src]+z_t[dst]))
        castw(P(pb + 2), (long)din * f, wbuf);
        gemm(xs0, LD0, wbuf, f, tmp_s, f, P(pb + 3), N_EDGE, din, 0);
        combine_edge_kernel<<<g1(Ef), 256, 0, stream>>>(tmp_s, zt, src, dst, xs_f, xs_h, N_EDGE, f);

        // conv_t (K=2 Hodge-Laguerre): x@W0 + (x - L x)@W1 + b, then BN+ReLU
        fill0(lap, Nf);
        spmm_kernel<<<g1((long)NNZ_T * f), 256, 0, stream>>>(ei_t, ei_t + NNZ_T, w_t, xt_f, lap, NNZ_T, f);
        subcast_kernel<<<g1(Nf), 256, 0, stream>>>(xt_f, lap, u_h, Nf);
        castw(P(pb + 8), 2L * f * f, wbuf);
        gemm(xt_h, f, wbuf,               f, tmp_t, f, P(pb + 9), N_NODE, f, 0);
        gemm(u_h,  f, wbuf + (long)f * f, f, tmp_t, f, nullptr,   N_NODE, f, 1);
        fill0(stats, 1024);
        colstat_kernel<<<g1(Nf), 256, 0, stream>>>(tmp_t, stats, N_NODE, f);
        bn_relu_kernel<<<g1(Nf), 256, 0, stream>>>(tmp_t, stats, P(pb + 10), P(pb + 11),
                                                   xt_f, xt0 + din, LD0, N_NODE, f);

        // conv_s
        fill0(lap, Ef);
        spmm_kernel<<<g1((long)NNZ_S * f), 256, 0, stream>>>(ei_s, ei_s + NNZ_S, w_s, xs_f, lap, NNZ_S, f);
        subcast_kernel<<<g1(Ef), 256, 0, stream>>>(xs_f, lap, u_h, Ef);
        castw(P(pb + 12), 2L * f * f, wbuf);
        gemm(xs_h, f, wbuf,               f, tmp_s, f, P(pb + 13), N_EDGE, f, 0);
        gemm(u_h,  f, wbuf + (long)f * f, f, tmp_s, f, nullptr,    N_EDGE, f, 1);
        fill0(stats, 1024);
        colstat_kernel<<<g1(Ef), 256, 0, stream>>>(tmp_s, stats, N_EDGE, f);
        bn_relu_kernel<<<g1(Ef), 256, 0, stream>>>(tmp_s, stats, P(pb + 14), P(pb + 15),
                                                   xs_f, xs0 + din, LD0, N_EDGE, f);
    }

    // ================= segment mean-pool + output linear =================
    fill0(pooled, (long)G_BATCH * 1024);
    fill0(cnt, 2 * G_BATCH);
    count_kernel<<<g1(N_EDGE), 256, 0, stream>>>(s_batch, cnt, N_EDGE);
    count_kernel<<<g1(N_NODE), 256, 0, stream>>>(n_batch, cnt + G_BATCH, N_NODE);
    pool_accum_kernel<<<g1((long)N_EDGE * 512), 256, 0, stream>>>(xs_f, s_batch, pooled, N_EDGE, 512, 0);
    pool_accum_kernel<<<g1((long)N_NODE * 512), 256, 0, stream>>>(xt_f, n_batch, pooled, N_NODE, 512, 512);
    out_kernel<<<g1(G_BATCH), 256, 0, stream>>>(pooled, cnt, P(146), P(147), (float*)d_out, G_BATCH);
}
